// MyRNN_53876069761545
// MI455X (gfx1250) — matrix operand order repaired and checked by reading, not compiled, not run on hardware
//
#include <hip/hip_runtime.h>

typedef __attribute__((ext_vector_type(16))) __bf16 v16bf;
typedef __attribute__((ext_vector_type(8)))  __bf16 v8bf;
typedef __attribute__((ext_vector_type(8)))  float  v8f;
typedef __attribute__((ext_vector_type(4)))  int    v4i;

#define T_STEPS 2048
#define IN_DIM  2048
#define H_DIM   4096
#define OUT_DIM 2048

// Padded LDS row strides (elements) -> conflict-free ds_load_b128 across 16 rows
#define GEMM_LDW 40    // 20 words/row: banks 20*r mod 64 all distinct for r=0..15
#define W_LDW    4104  // 2052 words/row: banks 4*r mod 64, disjoint 4-word groups

// ---------------------------------------------------------------------------
// gfx1250 async global->LDS copy path (ASYNCcnt), with portable fallback.
// ---------------------------------------------------------------------------
#if __has_builtin(__builtin_amdgcn_global_load_async_to_lds_b128) && \
    __has_builtin(__builtin_amdgcn_s_wait_asynccnt)
#define HAVE_ASYNC_LDS 1
#endif

__device__ inline void cp_b128(const unsigned short* g, unsigned short* l) {
#ifdef HAVE_ASYNC_LDS
    typedef __attribute__((address_space(1))) v4i* gp_t;
    typedef __attribute__((address_space(3))) v4i* lp_t;
    __builtin_amdgcn_global_load_async_to_lds_b128((gp_t)g, (lp_t)l, 0, 0);
#else
    *reinterpret_cast<uint4*>(l) = *reinterpret_cast<const uint4*>(g);
#endif
}

__device__ inline void async_wait0() {
#ifdef HAVE_ASYNC_LDS
    __builtin_amdgcn_s_wait_asynccnt(0);
#endif
}

__device__ inline v8bf ld_v8bf(const unsigned short* p) {
    union { uint4 u; v8bf v; } c;
    c.u = *reinterpret_cast<const uint4*>(p);
    return c.v;
}

__device__ inline v16bf make_frag(v8bf lo, v8bf hi) {
    return __builtin_shufflevector(lo, hi, 0, 1, 2, 3, 4, 5, 6, 7,
                                           8, 9, 10, 11, 12, 13, 14, 15);
}

__device__ inline v16bf ld_frag_a(const unsigned short* p, int half) {
    // A 16x32 bf16 fragment: 16 contiguous K values starting at half*16
    return make_frag(ld_v8bf(p + half * 16), ld_v8bf(p + half * 16 + 8));
}

__device__ inline v16bf ld_frag_b(const unsigned short* p) {
    // B 32x16 bf16 fragment: 16 contiguous K values starting at half*16
    return make_frag(ld_v8bf(p), ld_v8bf(p + 8));
}

__device__ inline unsigned f2bf(float v) {
    unsigned u = __float_as_uint(v);
    return (u + 0x7FFFu + ((u >> 16) & 1u)) >> 16;   // RNE
}

// ---------------------------------------------------------------------------
// fp32 -> bf16 conversion, grid-stride
// ---------------------------------------------------------------------------
__global__ void to_bf16_kernel(const float* __restrict__ in,
                               unsigned short* __restrict__ out, int n) {
    int i = blockIdx.x * blockDim.x + threadIdx.x;
    int stride = gridDim.x * blockDim.x;
    for (; i < n; i += stride)
        out[i] = (unsigned short)f2bf(in[i]);
}

// ---------------------------------------------------------------------------
// Phase 1: xp[T,H] = x @ wx_w^T + (wx_b + wh_b).  bf16 WMMA, f32 accumulate.
// Block = 8 waves (4x2); block tile 128(M) x 128(N); wave tile 32x64
// (2x4 accumulators = 8 WMMAs per K-chunk, 12 ds_load_b128).
// A (128x32) and B (128x32) tiles staged in LDS via async copies,
// double-buffered so copy(k+1) overlaps WMMA(k). All fragments are loaded
// before the WMMA chain so one dscnt wait covers 8 matrix ops.
// ---------------------------------------------------------------------------
__global__ __launch_bounds__(256) void gemm_xp_kernel(
    const unsigned short* __restrict__ xbf,   // [T, IN] bf16
    const unsigned short* __restrict__ wxbf,  // [H, IN] bf16
    const float* __restrict__ wx_b,
    const float* __restrict__ wh_b,
    float* __restrict__ xp) {                 // [T, H] f32
    __shared__ unsigned short lA[2][128 * GEMM_LDW];
    __shared__ unsigned short lB[2][128 * GEMM_LDW];

    const int tid  = threadIdx.x;
    const int lane = tid & 31;
    const int wave = tid >> 5;
    const int half = lane >> 4;
    const int l16  = lane & 15;
    const int wrow = wave >> 1;           // 0..3  -> 32-row group
    const int wcol = wave & 1;            // 0..1  -> 64-col group
    const int tBase = blockIdx.y * 128;
    const int j0    = blockIdx.x * 128;

    // Per-thread staging: 512 chunks per tile (128 rows x 4), 2 per thread.
    const int srow = tid >> 2;            // 0..63 (and srow+64)
    const int scol = (tid & 3) * 8;       // 0,8,16,24

    v8f acc[2][4] = { { {0.0f}, {0.0f}, {0.0f}, {0.0f} },
                      { {0.0f}, {0.0f}, {0.0f}, {0.0f} } };

    auto stage = [&](int buf, int k0) {
        cp_b128(xbf + (size_t)(tBase + srow) * IN_DIM + k0 + scol,
                &lA[buf][srow * GEMM_LDW + scol]);
        cp_b128(xbf + (size_t)(tBase + srow + 64) * IN_DIM + k0 + scol,
                &lA[buf][(srow + 64) * GEMM_LDW + scol]);
        cp_b128(wxbf + (size_t)(j0 + srow) * IN_DIM + k0 + scol,
                &lB[buf][srow * GEMM_LDW + scol]);
        cp_b128(wxbf + (size_t)(j0 + srow + 64) * IN_DIM + k0 + scol,
                &lB[buf][(srow + 64) * GEMM_LDW + scol]);
    };

    stage(0, 0);
    int buf = 0;
    for (int k0 = 0; k0 < IN_DIM; k0 += 32) {
        async_wait0();
        __syncthreads();
        if (k0 + 32 < IN_DIM) stage(buf ^ 1, k0 + 32);

        // Load ALL fragments first, then the 8-WMMA chain.
        v16bf a0 = ld_frag_a(&lA[buf][(wrow * 32 + l16) * GEMM_LDW], half);
        v16bf a1 = ld_frag_a(&lA[buf][(wrow * 32 + 16 + l16) * GEMM_LDW], half);
        v16bf b[4];
#pragma unroll
        for (int f = 0; f < 4; ++f)
            b[f] = ld_frag_b(
                &lB[buf][(wcol * 64 + f * 16 + l16) * GEMM_LDW + half * 16]);
#pragma unroll
        for (int f = 0; f < 4; ++f) {
            acc[0][f] = __builtin_amdgcn_wmma_f32_16x16x32_bf16(
                false, a0, false, b[f], (short)0, acc[0][f], false, false);
            acc[1][f] = __builtin_amdgcn_wmma_f32_16x16x32_bf16(
                false, a1, false, b[f], (short)0, acc[1][f], false, false);
        }
        buf ^= 1;
    }

#pragma unroll
    for (int i = 0; i < 2; ++i) {
#pragma unroll
        for (int f = 0; f < 4; ++f) {
            const int col = j0 + wcol * 64 + f * 16 + l16;
            const float bias = wx_b[col] + wh_b[col];
#pragma unroll
            for (int r = 0; r < 8; ++r) {
                const int row = tBase + wrow * 32 + i * 16 + half * 8 + r;
                xp[(size_t)row * H_DIM + col] = acc[i][f][r] + bias;
            }
        }
    }
}

// ---------------------------------------------------------------------------
// Phase 2: persistent recurrence. 256 blocks; block owns 16 h-outputs and
// keeps its 16x4096 bf16 slice of wh_w resident in LDS (padded rows,
// ~128 KB of the WGP's 320 KB) for all 2048 timesteps. Per step:
//   h -> bf16 image in LDS, GEMV via WMMA (h replicated across A rows,
//   software-pipelined frag loads), cross-wave LDS reduce, tanh,
//   publish slice, device-wide barrier.
// ---------------------------------------------------------------------------
__global__ __launch_bounds__(256, 1) void rnn_steps_kernel(
    const float* __restrict__ xp,             // [T, H], bias folded in
    const unsigned short* __restrict__ whbf,  // [H, H] bf16
    float* __restrict__ h0,
    float* __restrict__ h1,
    unsigned* bar) {
    __shared__ unsigned short wlds[16 * W_LDW];   // padded weight slice
    __shared__ unsigned short hbf[H_DIM];         // bf16 image of h (8 KB)
    __shared__ float red[8][16];

    const int tid  = threadIdx.x;
    const int lane = tid & 31;
    const int wave = tid >> 5;
    const int half = lane >> 4;
    const int l16  = lane & 15;
    const int n0   = blockIdx.x * 16;
    const unsigned nblk = gridDim.x;

    // Stage the 16-row weight slice once (async copies, padded LDS rows).
    for (int c = tid; c < 16 * (H_DIM / 8); c += 256) {
        const int r = c >> 9;               // 512 chunks per row
        const int off = (c & 511) * 8;
        cp_b128(whbf + (size_t)(n0 + r) * H_DIM + off,
                &wlds[r * W_LDW + off]);
    }
    async_wait0();
    __syncthreads();

    const unsigned short* pwrow = wlds + l16 * W_LDW + half * 16;

    for (int t = 0; t < T_STEPS; ++t) {
        const float* hin = (t & 1) ? h1 : h0;
        float* hout      = (t & 1) ? h0 : h1;

        // Stage h as bf16 into LDS (vectorized f32 loads, packed stores).
        for (int i = tid; i < H_DIM / 4; i += 256) {
            float4 f = reinterpret_cast<const float4*>(hin)[i];
            uint2 p;
            p.x = f2bf(f.x) | (f2bf(f.y) << 16);
            p.y = f2bf(f.z) | (f2bf(f.w) << 16);
            reinterpret_cast<uint2*>(hbf)[i] = p;
        }
        __syncthreads();

        // Software-pipelined WMMA GEMV: prefetch chunk c+8 during wmma(c).
        v8f acc = {0.0f};
        int k0 = wave * 32;
        v16bf a = ld_frag_a(hbf + k0, half);
        v16bf b = ld_frag_b(pwrow + k0);
        for (; k0 + 256 < H_DIM; k0 += 256) {
            v16bf an = ld_frag_a(hbf + k0 + 256, half);
            v16bf bn = ld_frag_b(pwrow + k0 + 256);
            acc = __builtin_amdgcn_wmma_f32_16x16x32_bf16(
                false, a, false, b, (short)0, acc, false, false);
            a = an; b = bn;
        }
        acc = __builtin_amdgcn_wmma_f32_16x16x32_bf16(
            false, a, false, b, (short)0, acc, false, false);

        // A rows replicated -> acc[0] (lanes 0-15) = partial for col n0+lane.
        if (lane < 16) red[wave][lane] = acc[0];
        __syncthreads();
        if (tid < 16) {
            float s = 0.0f;
#pragma unroll
            for (int w = 0; w < 8; ++w) s += red[w][tid];
            s += xp[(size_t)t * H_DIM + n0 + tid];
            hout[n0 + tid] = tanhf(s);
        }

        // Device-wide barrier between timesteps.
        __threadfence();
        __syncthreads();
        if (tid == 0) {
            unsigned gen = *(volatile unsigned*)&bar[1];
            if (atomicAdd(&bar[0], 1u) == nblk - 1u) {
                atomicExch(&bar[0], 0u);
                __threadfence();
                atomicAdd(&bar[1], 1u);
            } else {
                while (*(volatile unsigned*)&bar[1] == gen)
                    __builtin_amdgcn_s_sleep(2);
            }
        }
        __threadfence();
        __syncthreads();
    }
}

// ---------------------------------------------------------------------------
// Phase 3: out = sigmoid(hT @ wo_w^T + wo_b). Wave-per-output fp32 GEMV.
// ---------------------------------------------------------------------------
__global__ __launch_bounds__(256) void out_kernel(
    const float* __restrict__ h,
    const float* __restrict__ wo_w,
    const float* __restrict__ wo_b,
    float* __restrict__ out) {
    const int wave = threadIdx.x >> 5;
    const int lane = threadIdx.x & 31;
    const int j = blockIdx.x * 8 + wave;
    const float* row = wo_w + (size_t)j * H_DIM;
    float s = 0.0f;
    for (int k = lane; k < H_DIM; k += 32) s += h[k] * row[k];
#pragma unroll
    for (int off = 16; off > 0; off >>= 1) s += __shfl_xor(s, off, 32);
    if (lane == 0) out[j] = 1.0f / (1.0f + expf(-(s + wo_b[j])));
}

// ---------------------------------------------------------------------------
extern "C" void kernel_launch(void* const* d_in, const int* in_sizes, int n_in,
                              void* d_out, int out_size, void* d_ws, size_t ws_size,
                              hipStream_t stream) {
    const float* x    = (const float*)d_in[0];  // [T, IN]
    const float* wh_w = (const float*)d_in[1];  // [H, H]
    const float* wh_b = (const float*)d_in[2];  // [H]
    const float* wx_w = (const float*)d_in[3];  // [H, IN]
    const float* wx_b = (const float*)d_in[4];  // [H]
    const float* wo_w = (const float*)d_in[5];  // [OUT, H]
    const float* wo_b = (const float*)d_in[6];  // [OUT]
    float* out = (float*)d_out;

    char* ws = (char*)d_ws;
    unsigned*       bar  = (unsigned*)ws;                       // 2 x u32
    float*          h0   = (float*)(ws + 4096);                 // 16 KB
    float*          h1   = (float*)(ws + 20480);                // 16 KB
    float*          xp   = (float*)(ws + 40960);                // 32 MB
    unsigned short* xbf  = (unsigned short*)(ws + 40960 + 33554432);
    unsigned short* wxbf = xbf + (size_t)T_STEPS * IN_DIM;      // 16 MB
    unsigned short* whbf = wxbf + (size_t)H_DIM * IN_DIM;       // 32 MB

    (void)hipMemsetAsync(ws, 0, 40960, stream);   // barrier state + h buffers

    to_bf16_kernel<<<2048, 256, 0, stream>>>(x, xbf, T_STEPS * IN_DIM);
    to_bf16_kernel<<<2048, 256, 0, stream>>>(wx_w, wxbf, H_DIM * IN_DIM);
    to_bf16_kernel<<<4096, 256, 0, stream>>>(wh_w, whbf, H_DIM * H_DIM);

    gemm_xp_kernel<<<dim3(H_DIM / 128, T_STEPS / 128), 256, 0, stream>>>(
        xbf, wxbf, wx_b, wh_b, xp);

    rnn_steps_kernel<<<H_DIM / 16, 256, 0, stream>>>(xp, whbf, h0, h1, bar);

    out_kernel<<<OUT_DIM / 8, 256, 0, stream>>>(h0, wo_w, wo_b, out);
}
